// Memory_57638461112422
// MI455X (gfx1250) — matrix operand order repaired and checked by reading, not compiled, not run on hardware
//
#include <hip/hip_runtime.h>
#include <stdint.h>

#define Bq 32
#define Nq 1024
#define Wq 64
#define Rq 4
#define OUTq 256
#define IFq 471
#define EPSq 1e-6f

typedef __attribute__((ext_vector_type(16))) _Float16 v16h;
typedef __attribute__((ext_vector_type(8)))  float    v8f;
typedef __attribute__((ext_vector_type(4)))  float    f4;
typedef __attribute__((ext_vector_type(2)))  uint32_t u32x2;

union AFrag { v16h v; uint32_t u[8]; _Float16 h[16]; };
union CFrag { v8f  v; float f[8]; };
union Pack4 { u32x2 u2; _Float16 h[4]; };

__device__ __forceinline__ float sigmoidf_(float x){ return 1.f/(1.f+__expf(-x)); }
__device__ __forceinline__ float oneplusf_(float x){ return 1.f + (x>20.f ? x : log1pf(__expf(x))); }

// ---------------- K1: interface split ----------------
__global__ void k_iface(const float* __restrict__ itf,
                        float* wkey, float* wvec, float* evec,
                        float* fg, float* rstr, float* wstr, float* wgate, float* agate,
                        float* rkeys, float* rmode)
{
  int b = blockIdx.x, t = threadIdx.x;           // 128 threads
  const float* row = itf + b*IFq;
  for (int w = t; w < Wq; w += 128) {
    wkey[b*Wq+w] = row[w];
    wvec[b*Wq+w] = row[Wq + w];
    evec[b*Wq+w] = sigmoidf_(row[2*Wq + w]);
  }
  if (t < Rq) {
    fg[b*Rq+t]   = sigmoidf_(row[3*Wq + t]);
    rstr[b*Rq+t] = oneplusf_(row[3*Wq + Rq + t]);
  }
  if (t == 0) {
    wstr[b]  = oneplusf_(row[3*Wq + 2*Rq + 0]);
    wgate[b] = sigmoidf_(row[3*Wq + 2*Rq + 1]);
    agate[b] = sigmoidf_(row[3*Wq + 2*Rq + 2]);
  }
  const int ko = 3*Wq + 2*Rq + 3;
  for (int i = t; i < Rq*Wq; i += 128) rkeys[b*Rq*Wq + i] = row[ko + i];  // [b][r][w]
  const int mo = ko + Rq*Wq;
  if (t < Rq) {
    float m0 = row[mo+t*3+0], m1 = row[mo+t*3+1], m2 = row[mo+t*3+2];
    float mx = fmaxf(m0, fmaxf(m1, m2));
    float e0 = __expf(m0-mx), e1 = __expf(m1-mx), e2 = __expf(m2-mx);
    float inv = 1.f/(e0+e1+e2);
    rmode[b*3*Rq + 0*Rq + t] = e0*inv;
    rmode[b*3*Rq + 1*Rq + t] = e1*inv;
    rmode[b*3*Rq + 2*Rq + t] = e2*inv;
  }
}

// ---------------- K2: usage ----------------
__global__ void k_usage(const float* __restrict__ rw, const float* __restrict__ ww,
                        const float* __restrict__ uv, const float* __restrict__ fg,
                        float* __restrict__ usage_out)
{
  int gid = blockIdx.x*256 + threadIdx.x;        // B*N
  int b = gid / Nq;
  f4 r4 = *(const f4*)(rw + (size_t)gid*Rq);
  float ret = 1.f;
  #pragma unroll
  for (int r = 0; r < Rq; ++r) ret *= (1.f - fg[b*Rq+r]*r4[r]);
  float u = uv[gid], w = ww[gid];
  usage_out[gid] = (u + w - u*w) * ret;
}

// ---------------- K3: allocation weighting (bitonic sort + product scan) ----------------
__global__ void k_alloc(const float* __restrict__ usage, float* __restrict__ alloc)
{
  __shared__ float sv[Nq]; __shared__ int si[Nq]; __shared__ float cp[Nq];
  int b = blockIdx.x, t = threadIdx.x;           // 1024 threads
  sv[t] = EPSq + (1.f-EPSq)*usage[b*Nq + t]; si[t] = t;
  __syncthreads();
  for (int k = 2; k <= Nq; k <<= 1)
    for (int j = k>>1; j > 0; j >>= 1) {
      int p = t ^ j;
      if (p > t) {
        bool asc = ((t & k) == 0);
        float a = sv[t], c = sv[p];
        if ((a > c) == asc) { sv[t]=c; sv[p]=a; int ia=si[t]; si[t]=si[p]; si[p]=ia; }
      }
      __syncthreads();
    }
  cp[t] = sv[t]; __syncthreads();
  for (int off = 1; off < Nq; off <<= 1) {
    float x = cp[t];
    float y = (t >= off) ? cp[t-off] : 1.f;
    __syncthreads();
    cp[t] = x*y;
    __syncthreads();
  }
  float excl = (t == 0) ? 1.f : cp[t-1];
  alloc[b*Nq + si[t]] = (1.f - sv[t]) * excl;
}

// ---------------- K4: write content addressing + write_w + precedence ----------------
__global__ void k_write_addr(const float* __restrict__ mem, const float* __restrict__ wkey,
                             const float* __restrict__ wstr, const float* __restrict__ wgate,
                             const float* __restrict__ agate, const float* __restrict__ alloc,
                             const float* __restrict__ prec,
                             float* __restrict__ write_w_out, float* __restrict__ prec_out)
{
  __shared__ float key[Wq]; __shared__ float red[256]; __shared__ float bc;
  int b = blockIdx.x, t = threadIdx.x;           // 256 threads
  if (t < Wq) key[t] = wkey[b*Wq + t];
  __syncthreads();
  float ps = (t < Wq) ? key[t]*key[t] : 0.f;
  red[t] = ps; __syncthreads();
  for (int s = 128; s > 0; s >>= 1) { if (t < s) red[t] += red[t+s]; __syncthreads(); }
  if (t == 0) bc = sqrtf(red[0]);
  __syncthreads();
  float knorm = bc, str = wstr[b];
  float sim[4];
  #pragma unroll
  for (int q = 0; q < 4; ++q) {
    int n = t + q*256;
    const f4* mrow = (const f4*)(mem + ((size_t)b*Nq + n)*Wq);
    float dot = 0.f, msq = 0.f;
    #pragma unroll
    for (int wv = 0; wv < Wq/4; ++wv) {
      f4 m = mrow[wv];
      #pragma unroll
      for (int e = 0; e < 4; ++e) { dot += m[e]*key[wv*4+e]; msq += m[e]*m[e]; }
    }
    sim[q] = dot / (sqrtf(msq)*knorm + EPSq) * str;
  }
  float mx = fmaxf(fmaxf(sim[0],sim[1]), fmaxf(sim[2],sim[3]));
  __syncthreads();
  red[t] = mx; __syncthreads();
  for (int s = 128; s > 0; s >>= 1) { if (t < s) red[t] = fmaxf(red[t], red[t+s]); __syncthreads(); }
  if (t == 0) bc = red[0];
  __syncthreads();
  mx = bc;
  float es = 0.f;
  #pragma unroll
  for (int q = 0; q < 4; ++q) { sim[q] = __expf(sim[q]-mx); es += sim[q]; }
  __syncthreads();
  red[t] = es; __syncthreads();
  for (int s = 128; s > 0; s >>= 1) { if (t < s) red[t] += red[t+s]; __syncthreads(); }
  if (t == 0) bc = 1.f/red[0];
  __syncthreads();
  float inv = bc, wg = wgate[b], ag = agate[b];
  float wloc[4];
  #pragma unroll
  for (int q = 0; q < 4; ++q) {
    int n = t + q*256;
    float wa = sim[q]*inv;
    float ww = wg*((1.f-ag)*wa + ag*alloc[b*Nq + n]);
    wloc[q] = ww;
    write_w_out[b*Nq + n] = ww;
  }
  float s4 = wloc[0]+wloc[1]+wloc[2]+wloc[3];
  __syncthreads();
  red[t] = s4; __syncthreads();
  for (int s = 128; s > 0; s >>= 1) { if (t < s) red[t] += red[t+s]; __syncthreads(); }
  if (t == 0) bc = red[0];
  __syncthreads();
  float sw = bc;
  #pragma unroll
  for (int q = 0; q < 4; ++q) {
    int n = t + q*256;
    prec_out[b*Nq + n] = (1.f - sw)*prec[b*Nq + n] + wloc[q];
  }
}

// ---------------- K5: memory erase/add (b128 streams) ----------------
__global__ void k_memupd(const float* __restrict__ mem, const float* __restrict__ ww,
                         const float* __restrict__ evec, const float* __restrict__ wvec,
                         float* __restrict__ out)
{
  size_t v = (size_t)blockIdx.x*256 + threadIdx.x;     // over B*N*W/4
  size_t idx = v*4;
  size_t bn = idx / Wq; int w = (int)(idx % Wq); int b = (int)(bn / Nq);
  float wwv = ww[bn];
  f4 m = *(const f4*)(mem + idx);
  f4 e = *(const f4*)(evec + b*Wq + w);
  f4 a = *(const f4*)(wvec + b*Wq + w);
  f4 o;
  #pragma unroll
  for (int i = 0; i < 4; ++i) o[i] = m[i]*(1.f - wwv*e[i]) + wwv*a[i];
  *(f4*)(out + idx) = o;
}

// ---------------- K6: link update + fused forward (WMMA) / backward einsums ----------------
// One block per 16-row strip. b128 NT stream of link in/out, f16 tile double-buffered in
// LDS (1 barrier/chunk), B operand pre-staged once in WMMA layout for the whole batch.
__global__ void k_link(const float* __restrict__ link, const float* __restrict__ prec,
                       const float* __restrict__ rw, const float* __restrict__ write_w,
                       float* __restrict__ link_out, float* __restrict__ fwd,
                       float* __restrict__ bwd)
{
  __shared__ float wj[Nq]; __shared__ float pj[Nq];
  __shared__ _Float16 tileA[2][16*32];     // double-buffered link_new tile (f16)
  __shared__ _Float16 rwAll[16*Nq];        // B operand: [n][k], n>=R rows are zero (32 KB)
  __shared__ float rwS[16*Rq];             // strip of read_weights for backward
  int b = blockIdx.y, i0 = blockIdx.x*16, t = threadIdx.x;   // 128 threads = 4 waves
  for (int k = t; k < Nq; k += 128) { wj[k] = write_w[b*Nq+k]; pj[k] = prec[b*Nq+k]; }
  for (int k = t; k < Nq; k += 128) {
    f4 q = *(const f4*)(rw + ((size_t)b*Nq + k)*Rq);
    rwAll[0*Nq + k] = (_Float16)q[0];
    rwAll[1*Nq + k] = (_Float16)q[1];
    rwAll[2*Nq + k] = (_Float16)q[2];
    rwAll[3*Nq + k] = (_Float16)q[3];
  }
  for (int idx = 4*Nq + t; idx < 16*Nq; idx += 128) rwAll[idx] = (_Float16)0.f;
  if (t < 16*Rq) { int i = t>>2, r = t&3; rwS[t] = rw[((size_t)b*Nq + i0 + i)*Rq + r]; }
  __syncthreads();

  int lane = t & 31, wave = t >> 5;
  int ti = t >> 3, tj = (t & 7)*4;          // each thread: one float4 of the 16x32 tile
  float wi = wj[i0 + ti];
  CFrag c;
  #pragma unroll
  for (int g = 0; g < 8; ++g) c.f[g] = 0.f;

  int arow = lane & 15, abase = (lane < 16) ? 0 : 8;
  int bn   = lane & 15, bbase = (lane < 16) ? 0 : 8;
  const uint32_t* tb = (const uint32_t*)rwAll;

  for (int j0 = 0; j0 < Nq; j0 += 32) {
    int buf = (j0 >> 5) & 1;
    // compute float4 of link_new; NT b128 in/out; pack 4 halfs -> one b64 LDS store
    size_t g = ((size_t)b*Nq + (i0 + ti))*Nq + (j0 + tj);
    f4 lv = __builtin_nontemporal_load((const f4*)(link + g));
    f4 ln; Pack4 pk;
    #pragma unroll
    for (int e = 0; e < 4; ++e) {
      int jj = j0 + tj + e;
      float vv = (i0 + ti == jj) ? 0.f : (1.f - wi + wj[jj])*lv[e] + wi*pj[jj];
      ln[e] = vv; pk.h[e] = (_Float16)vv;
    }
    __builtin_nontemporal_store(ln, (f4*)(link_out + g));
    *(u32x2*)(&tileA[buf][ti*32 + tj]) = pk.u2;
    __syncthreads();
    if (wave == 0) {   // forward_w accumulate: C += A(16x32 tile) x B(32x16 rw)
      AFrag a, bb;
      const uint32_t* ta = (const uint32_t*)&tileA[buf][0];
      #pragma unroll
      for (int v = 0; v < 8; ++v) {
        int k = ((v < 4) ? 2*v : 16 + 2*(v-4)) + abase;
        a.u[v] = ta[(arow*32 + k) >> 1];
      }
      #pragma unroll
      for (int v = 0; v < 8; ++v) bb.u[v] = tb[(bn*Nq + j0 + ((v < 4) ? 2*v : 16 + 2*(v-4)) + bbase) >> 1];
      c.v = __builtin_amdgcn_wmma_f32_16x16x32_f16(false, a.v, false, bb.v,
                                                   (short)0, c.v, false, false);
    }
    { // backward partial: one (j,r) per thread, reduce over the 16 strip rows
      int j = t >> 2, r = t & 3;
      float s = 0.f;
      #pragma unroll
      for (int i = 0; i < 16; ++i) s += (float)tileA[buf][i*32 + j] * rwS[i*Rq + r];
      atomicAdd(&bwd[((size_t)b*Nq + j0 + j)*Rq + r], s);
    }
  }
  if (wave == 0 && (lane & 15) < Rq) {     // forward rows out (cols >= R are exact zeros)
    int r = lane & 15, mbase = (lane < 16) ? 0 : 8;
    #pragma unroll
    for (int g = 0; g < 8; ++g)
      fwd[((size_t)b*Nq + i0 + mbase + g)*Rq + r] = c.f[g];
  }
}

// ---------------- K7: read content addressing ----------------
__global__ void k_read_addr(const float* __restrict__ memu, const float* __restrict__ rkeys,
                            const float* __restrict__ rstr, float* __restrict__ raddr)
{
  __shared__ float keys[Rq][Wq]; __shared__ float red[256]; __shared__ float bc;
  __shared__ float knorm[Rq];
  int b = blockIdx.x, t = threadIdx.x;           // 256 threads
  for (int i = t; i < Rq*Wq; i += 256) keys[i/Wq][i%Wq] = rkeys[b*Rq*Wq + i];
  __syncthreads();
  if (t < Rq) {
    float s = 0.f;
    for (int w = 0; w < Wq; ++w) { float k = keys[t][w]; s += k*k; }
    knorm[t] = sqrtf(s);
  }
  __syncthreads();
  float sim[4][Rq];
  #pragma unroll
  for (int q = 0; q < 4; ++q) {
    int n = t + q*256;
    const f4* mrow = (const f4*)(memu + ((size_t)b*Nq + n)*Wq);
    float msq = 0.f, dot[Rq] = {0,0,0,0};
    #pragma unroll
    for (int wv = 0; wv < Wq/4; ++wv) {
      f4 m = mrow[wv];
      #pragma unroll
      for (int e = 0; e < 4; ++e) {
        msq += m[e]*m[e];
        #pragma unroll
        for (int r = 0; r < Rq; ++r) dot[r] += m[e]*keys[r][wv*4+e];
      }
    }
    float mn = sqrtf(msq);
    #pragma unroll
    for (int r = 0; r < Rq; ++r) sim[q][r] = dot[r]/(mn*knorm[r] + EPSq)*rstr[b*Rq + r];
  }
  for (int r = 0; r < Rq; ++r) {
    float mx = fmaxf(fmaxf(sim[0][r],sim[1][r]), fmaxf(sim[2][r],sim[3][r]));
    red[t] = mx; __syncthreads();
    for (int s = 128; s > 0; s >>= 1) { if (t < s) red[t] = fmaxf(red[t], red[t+s]); __syncthreads(); }
    if (t == 0) bc = red[0];
    __syncthreads();
    mx = bc;
    float es = 0.f;
    #pragma unroll
    for (int q = 0; q < 4; ++q) { sim[q][r] = __expf(sim[q][r]-mx); es += sim[q][r]; }
    __syncthreads();
    red[t] = es; __syncthreads();
    for (int s = 128; s > 0; s >>= 1) { if (t < s) red[t] += red[t+s]; __syncthreads(); }
    if (t == 0) bc = 1.f/red[0];
    __syncthreads();
    float inv = bc;
    #pragma unroll
    for (int q = 0; q < 4; ++q) {
      int n = t + q*256;
      raddr[((size_t)b*Nq + n)*Rq + r] = sim[q][r]*inv;
    }
    __syncthreads();
  }
}

// ---------------- K8: mix read weights (b128) ----------------
__global__ void k_read_w(const float* __restrict__ bwd, const float* __restrict__ raddr,
                         const float* __restrict__ fwd, const float* __restrict__ rmode,
                         float* __restrict__ read_w)
{
  int gid = blockIdx.x*256 + threadIdx.x;        // B*N, 4 r's per thread
  int b = gid / Nq;
  f4 bw = *(const f4*)(bwd   + (size_t)gid*Rq);
  f4 ra = *(const f4*)(raddr + (size_t)gid*Rq);
  f4 fw = *(const f4*)(fwd   + (size_t)gid*Rq);
  f4 m0 = *(const f4*)(rmode + b*12 + 0);
  f4 m1 = *(const f4*)(rmode + b*12 + 4);
  f4 m2 = *(const f4*)(rmode + b*12 + 8);
  f4 o;
  #pragma unroll
  for (int r = 0; r < 4; ++r) o[r] = bw[r]*m0[r] + ra[r]*m1[r] + fw[r]*m2[r];
  *(f4*)(read_w + (size_t)gid*Rq) = o;
}

// ---------------- K9: read vectors ----------------
__global__ void k_read_vec(const float* __restrict__ memu, const float* __restrict__ read_w,
                           float* __restrict__ rvec)
{
  __shared__ float part[4][Wq][Rq];
  int b = blockIdx.x, t = threadIdx.x;           // 256 threads
  int w = t & 63, g = t >> 6;
  f4 acc = {0.f, 0.f, 0.f, 0.f};
  for (int n = g; n < Nq; n += 4) {
    float m = memu[((size_t)b*Nq + n)*Wq + w];
    f4 rw4 = *(const f4*)(read_w + ((size_t)b*Nq + n)*Rq);
    #pragma unroll
    for (int r = 0; r < Rq; ++r) acc[r] += m*rw4[r];
  }
  #pragma unroll
  for (int r = 0; r < Rq; ++r) part[g][w][r] = acc[r];
  __syncthreads();
  if (g == 0) {
    #pragma unroll
    for (int r = 0; r < Rq; ++r)
      rvec[b*Wq*Rq + w*Rq + r] = part[0][w][r]+part[1][w][r]+part[2][w][r]+part[3][w][r];
  }
}

// ---------------- K10: output projection, WMMA GEMM 32x256x256 ----------------
__global__ void k_proj(const float* __restrict__ flat, const float* __restrict__ Wout,
                       const float* __restrict__ bout, float* __restrict__ out)
{
  int t = threadIdx.x;                           // 256 threads = 8 waves
  int lane = t & 31, wv = t >> 5;
  int tile = blockIdx.x*8 + wv;                  // 0..31: 2 M-tiles x 16 N-tiles
  int mt = tile & 1, nt = tile >> 1;
  CFrag c;
  #pragma unroll
  for (int g = 0; g < 8; ++g) c.f[g] = 0.f;
  int arow  = mt*16 + (lane & 15);
  int abase = (lane < 16) ? 0 : 8;
  int bcol  = nt*16 + (lane & 15);
  int bbase = (lane < 16) ? 0 : 8;
  for (int k0 = 0; k0 < OUTq; k0 += 32) {
    AFrag a, bb;
    #pragma unroll
    for (int v = 0; v < 8; ++v) {
      int k = k0 + ((v < 4) ? 2*v : 16 + 2*(v-4)) + abase;
      a.h[2*v]   = (_Float16)flat[arow*OUTq + k];
      a.h[2*v+1] = (_Float16)flat[arow*OUTq + k + 1];
    }
    #pragma unroll
    for (int v = 0; v < 8; ++v) {
      int k = k0 + ((v < 4) ? 2*v : 16 + 2*(v-4)) + bbase;
      bb.h[2*v]   = (_Float16)Wout[(size_t)k*OUTq + bcol];
      bb.h[2*v+1] = (_Float16)Wout[(size_t)(k+1)*OUTq + bcol];
    }
    c.v = __builtin_amdgcn_wmma_f32_16x16x32_f16(false, a.v, false, bb.v,
                                                 (short)0, c.v, false, false);
  }
  int col   = nt*16 + (lane & 15);
  int rbase = mt*16 + ((lane < 16) ? 0 : 8);
  float bv = bout[col];
  #pragma unroll
  for (int g = 0; g < 8; ++g) out[(rbase+g)*OUTq + col] = c.f[g] + bv;
}

// ---------------- K0: zero scratch (b128) ----------------
__global__ void k_zero(float* __restrict__ p, int n4)
{
  int i = blockIdx.x*256 + threadIdx.x;
  if (i < n4) { f4 z = {0.f,0.f,0.f,0.f}; *(f4*)(p + (size_t)i*4) = z; }
}

extern "C" void kernel_launch(void* const* d_in, const int* in_sizes, int n_in,
                              void* d_out, int out_size, void* d_ws, size_t ws_size,
                              hipStream_t stream)
{
  const float* itf  = (const float*)d_in[0];
  const float* mem  = (const float*)d_in[1];
  const float* rw   = (const float*)d_in[2];
  const float* wwin = (const float*)d_in[3];
  const float* uv   = (const float*)d_in[4];
  const float* prec = (const float*)d_in[5];
  const float* link = (const float*)d_in[6];
  const float* Wout = (const float*)d_in[7];
  const float* bout = (const float*)d_in[8];

  float* out    = (float*)d_out;           // outputs, flat in return order
  float* o_mo    = out;                                   // [B,OUT]
  float* o_memu  = o_mo    + Bq*OUTq;                     // [B,N,W]
  float* o_readw = o_memu  + (size_t)Bq*Nq*Wq;            // [B,N,R]
  float* o_ww    = o_readw + (size_t)Bq*Nq*Rq;            // [B,N,1]
  float* o_rvec  = o_ww    + Bq*Nq;                       // [B,W,R]
  float* o_usage = o_rvec  + Bq*Wq*Rq;                    // [B,N]
  float* o_prec  = o_usage + Bq*Nq;                       // [B,N]
  float* o_link  = o_prec  + Bq*Nq;                       // [B,N,N]

  float* ws = (float*)d_ws;
  float* w_wkey  = ws;               ws += Bq*Wq;
  float* w_wvec  = ws;               ws += Bq*Wq;
  float* w_evec  = ws;               ws += Bq*Wq;
  float* w_fg    = ws;               ws += Bq*Rq;
  float* w_rstr  = ws;               ws += Bq*Rq;
  float* w_wstr  = ws;               ws += Bq;
  float* w_wgate = ws;               ws += Bq;
  float* w_agate = ws;               ws += Bq;
  float* w_rkeys = ws;               ws += Bq*Rq*Wq;
  float* w_rmode = ws;               ws += Bq*3*Rq;
  float* w_alloc = ws;               ws += Bq*Nq;
  float* w_fwd   = ws;               ws += Bq*Nq*Rq;
  float* w_bwd   = ws;               ws += Bq*Nq*Rq;
  float* w_raddr = ws;               ws += Bq*Nq*Rq;

  // K0: backward accumulator must start at zero each call (atomics)
  k_zero<<<(Bq*Nq*Rq/4 + 255)/256, 256, 0, stream>>>(w_bwd, Bq*Nq*Rq/4);
  // K1
  k_iface<<<Bq, 128, 0, stream>>>(itf, w_wkey, w_wvec, w_evec, w_fg, w_rstr,
                                  w_wstr, w_wgate, w_agate, w_rkeys, w_rmode);
  // K2
  k_usage<<<(Bq*Nq)/256, 256, 0, stream>>>(rw, wwin, uv, w_fg, o_usage);
  // K3
  k_alloc<<<Bq, 1024, 0, stream>>>(o_usage, w_alloc);
  // K4
  k_write_addr<<<Bq, 256, 0, stream>>>(mem, w_wkey, w_wstr, w_wgate, w_agate,
                                       w_alloc, prec, o_ww, o_prec);
  // K5
  k_memupd<<<(Bq*Nq*Wq/4)/256, 256, 0, stream>>>(mem, o_ww, w_evec, w_wvec, o_memu);
  // K6: big link stream + fused einsums (WMMA forward, atomic backward)
  k_link<<<dim3(Nq/16, Bq), 128, 0, stream>>>(link, prec, rw, o_ww,
                                              o_link, w_fwd, w_bwd);
  // K7
  k_read_addr<<<Bq, 256, 0, stream>>>(o_memu, w_rkeys, w_rstr, w_raddr);
  // K8
  k_read_w<<<(Bq*Nq)/256, 256, 0, stream>>>(w_bwd, w_raddr, w_fwd, w_rmode, o_readw);
  // K9
  k_read_vec<<<Bq, 256, 0, stream>>>(o_memu, o_readw, o_rvec);
  // K10: WMMA projection
  k_proj<<<4, 256, 0, stream>>>(o_rvec, Wout, bout, o_mo);
}
